// R2IR_3487513445074
// MI455X (gfx1250) — matrix operand order repaired and checked by reading, not compiled, not run on hardware
//
#include <hip/hip_runtime.h>
#include <hip/hip_bf16.h>
#include <math.h>

// ---------------------------------------------------------------------------
// MI455X (gfx1250) implementation of the linear-attention transformer block.
// All heavy GEMMs run on v_wmma_f32_16x16x32_f16 (wave32, 8 waves/block,
// 128x64 block tile, K-step 32). Global->LDS staging uses the gfx1250 async
// load-to-LDS path, double-buffered so next-tile loads fly under the WMMAs.
// ---------------------------------------------------------------------------

typedef _Float16 h8   __attribute__((ext_vector_type(8)));
typedef _Float16 v16h __attribute__((ext_vector_type(16)));
typedef float    v8f  __attribute__((ext_vector_type(8)));
typedef int      v4i  __attribute__((vector_size(16)));

#define E_DIM  1024
#define N_TOK  4096
#define B_SZ   2
#define NHEAD  16
#define DHEAD  64
#define FF_DIM 4096
#define MTOT   8192   /* B_SZ * N_TOK */

// ------------------------- async-to-LDS feature probe ----------------------
#if defined(__HIP_DEVICE_COMPILE__) && defined(__has_builtin)
#if __has_builtin(__builtin_amdgcn_global_load_async_to_lds_b128)
#define HAS_ASYNC_LDS 1
#endif
#if __has_builtin(__builtin_amdgcn_s_wait_asynccnt)
#define HAS_WAIT_ASYNC 1
#endif
#endif
#ifndef HAS_ASYNC_LDS
#define HAS_ASYNC_LDS 0
#endif
#ifndef HAS_WAIT_ASYNC
#define HAS_WAIT_ASYNC 0
#endif

typedef __attribute__((address_space(1))) v4i* gv4i_p;   // global v4i32*
typedef __attribute__((address_space(3))) v4i* lv4i_p;   // LDS v4i32*

// copy 8 halves (16B) global -> LDS
__device__ __forceinline__ void cp8(const _Float16* g, _Float16* l) {
#if HAS_ASYNC_LDS
  __builtin_amdgcn_global_load_async_to_lds_b128(
      (gv4i_p)(void*)g, (lv4i_p)(void*)l, 0, 0);
#else
  *(h8*)l = *(const h8*)g;
#endif
}

// wait until at most N async ops outstanding (async loads complete in order)
__device__ __forceinline__ void async_wait0() {
#if HAS_ASYNC_LDS
#if HAS_WAIT_ASYNC
  __builtin_amdgcn_s_wait_asynccnt(0);
#else
  asm volatile("s_wait_asynccnt 0" ::: "memory");
#endif
#endif
}
__device__ __forceinline__ void async_wait3() {
#if HAS_ASYNC_LDS
#if HAS_WAIT_ASYNC
  __builtin_amdgcn_s_wait_asynccnt(3);
#else
  asm volatile("s_wait_asynccnt 3" ::: "memory");
#endif
#endif
}

// --------------------------- fragment loaders ------------------------------
// A fragment (16x32 f16): lane row = lane&15, k-base = (lane>>4)*8;
// halves s[0..7] = K kb..kb+7, s[8..15] = K kb+16..kb+23  -> two b128 loads.
__device__ __forceinline__ v16h frag_a(const _Float16* tile, int rowBase, int lane,
                                       int ldp) {
  const _Float16* p = tile + (rowBase + (lane & 15)) * ldp + ((lane >> 4) * 8);
  h8 lo = *(const h8*)p;
  h8 hi = *(const h8*)(p + 16);
  return __builtin_shufflevector(lo, hi, 0,1,2,3,4,5,6,7,8,9,10,11,12,13,14,15);
}
// B fragment (32x16 f16) from transposed LDS tile Bt[col][k]:
// lane col = lane&15, k-base = (lane>>4)*8; s[0..7] = K kb..kb+7, s[8..15] = K kb+16..kb+23.
__device__ __forceinline__ v16h frag_b(const _Float16* tile, int colBase, int lane,
                                       int ldp) {
  const _Float16* p = tile + (colBase + (lane & 15)) * ldp + ((lane >> 4) * 8);
  h8 lo = *(const h8*)p;
  h8 hi = *(const h8*)(p + 16);
  return __builtin_shufflevector(lo, hi, 0,1,2,3,4,5,6,7,8,9,10,11,12,13,14,15);
}

// ------------------------------ generic GEMM -------------------------------
// C[M,N] = A[M,K] (row-major f16) x Bc[N,K] (row-major f16 = B col-major)
// epilogue modes (template):
//   0: outF = acc + bias
//   1: outH = f16(acc + bias)
//   2: outH = f16(acc / (den + 1e-8))              (linear-attention num/den)
//   3: outF = add32 + (acc + bias) * scal[col]     (residual branches)
//   4: outH = f16(silu(acc + bias))                (FFN hidden)
#define BM 128
#define BN 64
#define BK 32
#define LDA_P 48
#define LDB_P 48

template <int MODE>
__global__ __launch_bounds__(256) void k_gemm(
    const _Float16* __restrict__ A, const _Float16* __restrict__ Bc,
    float* __restrict__ outF, _Float16* __restrict__ outH,
    const float* __restrict__ bias, const float* __restrict__ scal,
    const float* __restrict__ add32, const float* __restrict__ den,
    int M, int N, int K, int lda, int ldb, int ldc,
    long long aSo, long long aSi, long long bSo, long long bSi,
    long long cSo, long long cSi, int innerCount)
{
  __shared__ _Float16 lsA[2][BM * LDA_P];
  __shared__ _Float16 lsB[2][BN * LDB_P];

  const int z  = blockIdx.z;
  const int zo = z / innerCount;
  const int zi = z - zo * innerCount;
  const _Float16* Ab = A + zo * aSo + zi * aSi;
  const _Float16* Bb = Bc + zo * bSo + zi * bSi;
  const long long cOff = zo * cSo + zi * cSi;

  const int m0 = blockIdx.x * BM;
  const int n0 = blockIdx.y * BN;
  const int t = threadIdx.x, lane = t & 31, wave = t >> 5;
  const int wm = wave >> 1, wn = wave & 1;          // 4x2 wave grid
  const int arow = t >> 1, aoff = (t & 1) * 16;     // A tile fill
  const int brow = t >> 2, boff = (t & 3) * 8;      // B tile fill

  // per-iteration each thread issues 3 16B copies (2 for A, 1 for B)
  auto issue = [&](int k0, int buf) {
    cp8(Ab + (long long)(m0 + arow) * lda + k0 + aoff,     &lsA[buf][arow * LDA_P + aoff]);
    cp8(Ab + (long long)(m0 + arow) * lda + k0 + aoff + 8, &lsA[buf][arow * LDA_P + aoff + 8]);
    cp8(Bb + (long long)(n0 + brow) * ldb + k0 + boff,     &lsB[buf][brow * LDB_P + boff]);
  };

  v8f acc[2][2] = {};

  issue(0, 0);
  int cur = 0;
  for (int k0 = 0; k0 < K; k0 += BK) {
    const bool hasNext = (k0 + BK) < K;
    if (hasNext) issue(k0 + BK, cur ^ 1);   // prefetch next tile under compute
    if (hasNext) async_wait3();             // current tile's 3 copies done
    else         async_wait0();
    __syncthreads();

    v16h af[2], bf[2];
#pragma unroll
    for (int i = 0; i < 2; i++) af[i] = frag_a(lsA[cur], wm * 32 + i * 16, lane, LDA_P);
#pragma unroll
    for (int j = 0; j < 2; j++) bf[j] = frag_b(lsB[cur], wn * 32 + j * 16, lane, LDB_P);
#pragma unroll
    for (int i = 0; i < 2; i++)
#pragma unroll
      for (int j = 0; j < 2; j++)
        acc[i][j] = __builtin_amdgcn_wmma_f32_16x16x32_f16(
            false, af[i], false, bf[j], (short)0, acc[i][j], false, false);

    __syncthreads();   // all waves done reading 'cur' before it is refilled
    cur ^= 1;
  }

  // epilogue: lane holds col = lane&15, rows = (lane>>4)*8 + v
#pragma unroll
  for (int i = 0; i < 2; i++)
#pragma unroll
    for (int j = 0; j < 2; j++) {
      const int col = n0 + wn * 32 + j * 16 + (lane & 15);
      const int rb  = m0 + wm * 32 + i * 16 + ((lane >> 4) * 8);
#pragma unroll
      for (int v = 0; v < 8; v++) {
        const int m = rb + v;
        const float a = acc[i][j][v];
        const long long idx = cOff + (long long)m * ldc + col;
        if (MODE == 2) {
          const float d = den[(long long)zo * (N_TOK * NHEAD) + zi +
                              (long long)m * NHEAD] + 1e-8f;
          outH[idx] = (_Float16)(a / d);
        } else {
          const float r = a + bias[col];
          if (MODE == 0)      outF[idx] = r;
          else if (MODE == 1) outH[idx] = (_Float16)r;
          else if (MODE == 3) outF[idx] = add32[idx] + r * scal[col];
          else {              // 4: silu
            float s = r / (1.0f + __expf(-r));
            outH[idx] = (_Float16)s;
          }
        }
      }
    }
}

// --------------------- kv_sum kernel (per b,h 64x64 GEMM) ------------------
// kvT[z][e][d] = f16( sum_n k[b,n,h,d] * v[b,n,h,e] )   (z = b*NHEAD + h)
__global__ __launch_bounds__(128) void k_kv(const _Float16* __restrict__ k16,
                                            const _Float16* __restrict__ v16,
                                            _Float16* __restrict__ kvT)
{
  __shared__ _Float16 lsK[64 * 48];   // kT[d][n]
  __shared__ _Float16 lsV[64 * 48];   // vT[e][n]
  const int z = blockIdx.x, b = z >> 4, h = z & 15;
  const _Float16* Kb = k16 + (long long)b * N_TOK * E_DIM + h * DHEAD;
  const _Float16* Vb = v16 + (long long)b * N_TOK * E_DIM + h * DHEAD;
  const int t = threadIdx.x, lane = t & 31, wave = t >> 5;  // 4 waves
  const int wm = wave >> 1, wn = wave & 1;
  const int nl = t >> 2, d0 = (t & 3) * 16;

  v8f acc[2][2] = {};

  for (int n0 = 0; n0 < N_TOK; n0 += 32) {
    const h8* gk = (const h8*)(Kb + (long long)(n0 + nl) * E_DIM + d0);
    h8 ka = gk[0], kb2 = gk[1];
    const h8* gv = (const h8*)(Vb + (long long)(n0 + nl) * E_DIM + d0);
    h8 va = gv[0], vb2 = gv[1];
    __syncthreads();
#pragma unroll
    for (int jj = 0; jj < 8; jj++) {
      lsK[(d0 + jj)     * 48 + nl] = ka[jj];
      lsK[(d0 + 8 + jj) * 48 + nl] = kb2[jj];
      lsV[(d0 + jj)     * 48 + nl] = va[jj];
      lsV[(d0 + 8 + jj) * 48 + nl] = vb2[jj];
    }
    __syncthreads();

    v16h af[2], bf[2];
#pragma unroll
    for (int i = 0; i < 2; i++) af[i] = frag_a(lsK, wm * 32 + i * 16, lane, 48);
#pragma unroll
    for (int j = 0; j < 2; j++) bf[j] = frag_b(lsV, wn * 32 + j * 16, lane, 48);
#pragma unroll
    for (int i = 0; i < 2; i++)
#pragma unroll
      for (int j = 0; j < 2; j++)
        acc[i][j] = __builtin_amdgcn_wmma_f32_16x16x32_f16(
            false, af[i], false, bf[j], (short)0, acc[i][j], false, false);
  }

  _Float16* outp = kvT + (long long)z * (DHEAD * DHEAD);
#pragma unroll
  for (int i = 0; i < 2; i++)
#pragma unroll
    for (int j = 0; j < 2; j++) {
      const int e = wn * 32 + j * 16 + (lane & 15);
      const int db = wm * 32 + i * 16 + ((lane >> 4) * 8);
#pragma unroll
      for (int v = 0; v < 8; v++)
        outp[e * DHEAD + db + v] = (_Float16)acc[i][j][v];
    }
}

// ----------------------- elementwise / reduction kernels -------------------
__global__ __launch_bounds__(256) void k_w2h(const float* __restrict__ x,
                                             _Float16* __restrict__ y, int n) {
  int i = blockIdx.x * 256 + threadIdx.x;
  if (i < n) y[i] = (_Float16)x[i];
}

// img [B,E,N] f32 -> tokens [B,N,E] f32
__global__ __launch_bounds__(256) void k_img2tok_f32(const float* __restrict__ img,
                                                     float* __restrict__ tok) {
  __shared__ float tile[32][33];
  const int b = blockIdx.z, n0 = blockIdx.x * 32, e0 = blockIdx.y * 32;
  const int tx = threadIdx.x & 31, ty = threadIdx.x >> 5;
#pragma unroll
  for (int r = 0; r < 4; r++) {
    int e = ty + r * 8;
    tile[e][tx] = img[(long long)(b * E_DIM + e0 + e) * N_TOK + n0 + tx];
  }
  __syncthreads();
#pragma unroll
  for (int r = 0; r < 4; r++) {
    int n = ty + r * 8;
    tok[(long long)(b * N_TOK + n0 + n) * E_DIM + e0 + tx] = tile[tx][n];
  }
}

// img [B,E,N] f32 -> tokens [B,N,E] f16
__global__ __launch_bounds__(256) void k_img2tok_f16(const float* __restrict__ img,
                                                     _Float16* __restrict__ tok) {
  __shared__ float tile[32][33];
  const int b = blockIdx.z, n0 = blockIdx.x * 32, e0 = blockIdx.y * 32;
  const int tx = threadIdx.x & 31, ty = threadIdx.x >> 5;
#pragma unroll
  for (int r = 0; r < 4; r++) {
    int e = ty + r * 8;
    tile[e][tx] = img[(long long)(b * E_DIM + e0 + e) * N_TOK + n0 + tx];
  }
  __syncthreads();
#pragma unroll
  for (int r = 0; r < 4; r++) {
    int n = ty + r * 8;
    tok[(long long)(b * N_TOK + n0 + n) * E_DIM + e0 + tx] = (_Float16)tile[tx][n];
  }
}

// gx[b,e] = sqrt(sum_n X[b,n,e]^2)
__global__ __launch_bounds__(256) void k_grn_gx(const float* __restrict__ X,
                                                float* __restrict__ gx) {
  const int i = blockIdx.x * 256 + threadIdx.x;   // B*E = 2048
  const int b = i >> 10, e = i & (E_DIM - 1);
  const float* p = X + (long long)b * N_TOK * E_DIM + e;
  float s = 0.0f;
  for (int n = 0; n < N_TOK; n++) { float v = p[(long long)n * E_DIM]; s += v * v; }
  gx[i] = sqrtf(s);
}

// nx[b,e] = gx[b,e] / (mean_e gx[b,:] + 1e-6)
__global__ __launch_bounds__(256) void k_grn_nx(const float* __restrict__ gx,
                                                float* __restrict__ nx) {
  __shared__ float red[256];
  const int b = blockIdx.x, t = threadIdx.x;
  float s = gx[b * E_DIM + t] + gx[b * E_DIM + t + 256] +
            gx[b * E_DIM + t + 512] + gx[b * E_DIM + t + 768];
  red[t] = s;
  __syncthreads();
  for (int o = 128; o > 0; o >>= 1) {
    if (t < o) red[t] += red[t + o];
    __syncthreads();
  }
  const float inv = 1.0f / (red[0] * (1.0f / E_DIM) + 1e-6f);
  for (int j = t; j < E_DIM; j += 256) nx[b * E_DIM + j] = gx[b * E_DIM + j] * inv;
}

// out16[b,n,e] = f16( (1+gamma_e) * x * nx[b,e] + beta_e + x )
__global__ __launch_bounds__(256) void k_grn_apply(
    const float* __restrict__ X, const float* __restrict__ nx,
    const float* __restrict__ gam, const float* __restrict__ bet,
    _Float16* __restrict__ out) {
  const long long i = (long long)blockIdx.x * 256 + threadIdx.x;  // B*N*E
  const int e = (int)(i & (E_DIM - 1));
  const int b = (int)(i >> 22);   // N_TOK*E_DIM = 2^22
  const float x = X[i];
  const float f = nx[b * E_DIM + e];
  out[i] = (_Float16)((1.0f + gam[e]) * (x * f) + bet[e] + x);
}

// per-row LayerNorm over E=1024 (one wave per row), optional elu(x)+1
__global__ __launch_bounds__(256) void k_ln(const float* __restrict__ in,
                                            const float* __restrict__ w,
                                            const float* __restrict__ bb,
                                            _Float16* __restrict__ out, int elu) {
  const int row = blockIdx.x * 8 + (threadIdx.x >> 5);
  const int lane = threadIdx.x & 31;
  const float* p = in + (long long)row * E_DIM;
  float s = 0.0f, sq = 0.0f;
  for (int e = lane; e < E_DIM; e += 32) { float x = p[e]; s += x; sq += x * x; }
  for (int o = 16; o > 0; o >>= 1) { s += __shfl_xor(s, o); sq += __shfl_xor(sq, o); }
  const float mean = s * (1.0f / E_DIM);
  const float var  = sq * (1.0f / E_DIM) - mean * mean;
  const float rs   = rsqrtf(var + 1e-5f);
  for (int e = lane; e < E_DIM; e += 32) {
    float y = (p[e] - mean) * rs * w[e] + bb[e];
    if (elu) y = (y > 0.0f ? y : expm1f(y)) + 1.0f;
    out[(long long)row * E_DIM + e] = (_Float16)y;
  }
}

// ks[b,h,d] = sum_n k16[b,n,h,d]
__global__ __launch_bounds__(256) void k_ksum(const _Float16* __restrict__ k16,
                                              float* __restrict__ ks) {
  const int i = blockIdx.x * 256 + threadIdx.x;   // B*E = 2048
  const int b = i >> 10, hd = i & (E_DIM - 1);
  const _Float16* p = k16 + (long long)b * N_TOK * E_DIM + hd;
  float s = 0.0f;
  for (int n = 0; n < N_TOK; n++) s += (float)p[(long long)n * E_DIM];
  ks[i] = s;
}

// den[(b*N+n)*H+h] = sum_d q16[b,n,h,d]*ks[b,h,d]
__global__ __launch_bounds__(256) void k_den(const _Float16* __restrict__ q16,
                                             const float* __restrict__ ks,
                                             float* __restrict__ den) {
  const long long i = (long long)blockIdx.x * 256 + threadIdx.x;  // B*N*H
  const int h = (int)(i & (NHEAD - 1));
  const long long bn = i >> 4;
  const int b = (int)(bn >> 12);
  const _Float16* p = q16 + bn * E_DIM + h * DHEAD;
  const float* kp = ks + b * E_DIM + h * DHEAD;
  float s = 0.0f;
#pragma unroll 8
  for (int d = 0; d < DHEAD; d++) s += (float)p[d] * kp[d];
  den[i] = s;
}

// out[b,e,n] = qimg[b,e,n] + F[b,n,e] * fs[e]
__global__ __launch_bounds__(256) void k_final(const float* __restrict__ F,
                                               const float* __restrict__ qimg,
                                               const float* __restrict__ fs,
                                               float* __restrict__ out) {
  __shared__ float tile[32][33];
  const int b = blockIdx.z, n0 = blockIdx.x * 32, e0 = blockIdx.y * 32;
  const int tx = threadIdx.x & 31, ty = threadIdx.x >> 5;
#pragma unroll
  for (int r = 0; r < 4; r++) {
    int n = ty + r * 8;
    tile[n][tx] = F[(long long)(b * N_TOK + n0 + n) * E_DIM + e0 + tx];
  }
  __syncthreads();
#pragma unroll
  for (int r = 0; r < 4; r++) {
    int e = ty + r * 8;
    long long gi = (long long)(b * E_DIM + e0 + e) * N_TOK + n0 + tx;
    out[gi] = qimg[gi] + tile[tx][e] * fs[e0 + e];
  }
}

// ------------------------------- launcher ----------------------------------
template <int MODE>
static void launch_gemm(hipStream_t stream,
                        const _Float16* A, const _Float16* Bc, float* oF,
                        _Float16* oH, const float* bias, const float* scal,
                        const float* add32, const float* dn,
                        int M, int N, int K, int lda, int ldb, int ldc,
                        long long aSo, long long aSi, long long bSo,
                        long long bSi, long long cSo, long long cSi,
                        int innerCount, int batches) {
  dim3 g(M / BM, N / BN, batches);
  k_gemm<MODE><<<g, 256, 0, stream>>>(A, Bc, oF, oH, bias, scal, add32, dn,
                                      M, N, K, lda, ldb, ldc,
                                      aSo, aSi, bSo, bSi, cSo, cSi, innerCount);
}

extern "C" void kernel_launch(void* const* d_in, const int* in_sizes, int n_in,
                              void* d_out, int out_size, void* d_ws, size_t ws_size,
                              hipStream_t stream) {
  (void)in_sizes; (void)n_in; (void)out_size; (void)ws_size;
  const float* q_img = (const float*)d_in[0];
  const float* k_img = (const float*)d_in[1];
  const float* v_img = (const float*)d_in[2];
  const float* g1g = (const float*)d_in[3];
  const float* g1b = (const float*)d_in[4];
  const float* q_w = (const float*)d_in[5];
  const float* q_b = (const float*)d_in[6];
  const float* k_w = (const float*)d_in[7];
  const float* k_b = (const float*)d_in[8];
  const float* v_w = (const float*)d_in[9];
  const float* v_b = (const float*)d_in[10];
  const float* o_w = (const float*)d_in[11];
  const float* o_b = (const float*)d_in[12];
  const float* lnq_w = (const float*)d_in[13];
  const float* lnq_b = (const float*)d_in[14];
  const float* lnk_w = (const float*)d_in[15];
  const float* lnk_b = (const float*)d_in[16];
  const float* attn_s = (const float*)d_in[17];
  const float* g2g = (const float*)d_in[18];
  const float* g2b = (const float*)d_in[19];
  const float* w1  = (const float*)d_in[20];
  const float* b1  = (const float*)d_in[21];
  const float* w2  = (const float*)d_in[22];
  const float* b2  = (const float*)d_in[23];
  const float* ffn_s = (const float*)d_in[24];
  const float* fin_s = (const float*)d_in[25];
  float* out = (float*)d_out;

  // ---- workspace layout (~281 MB) ----
  char* ws = (char*)d_ws;
  size_t off = 0;
  auto alloc = [&](size_t bytes) {
    char* p = ws + off;
    off += (bytes + 255) & ~(size_t)255;
    return p;
  };
  const size_t SZ_EE  = (size_t)E_DIM * E_DIM;
  const size_t SZ_TOK = (size_t)MTOT * E_DIM;
  const size_t SZ_FF  = (size_t)MTOT * FF_DIM;

  _Float16* Wq = (_Float16*)alloc(SZ_EE * 2);
  _Float16* Wk = (_Float16*)alloc(SZ_EE * 2);
  _Float16* Wv = (_Float16*)alloc(SZ_EE * 2);
  _Float16* Wo = (_Float16*)alloc(SZ_EE * 2);
  _Float16* W1 = (_Float16*)alloc((size_t)FF_DIM * E_DIM * 2);
  _Float16* W2 = (_Float16*)alloc((size_t)E_DIM * FF_DIM * 2);
  float*    X    = (float*)alloc(SZ_TOK * 4);
  _Float16* Qf   = (_Float16*)alloc(SZ_TOK * 2);
  _Float16* Kf   = (_Float16*)alloc(SZ_TOK * 2);
  _Float16* Vf   = (_Float16*)alloc(SZ_TOK * 2);
  float*    tmp32 = (float*)alloc(SZ_TOK * 4);
  _Float16* q16  = (_Float16*)alloc(SZ_TOK * 2);
  _Float16* k16  = (_Float16*)alloc(SZ_TOK * 2);
  _Float16* v16  = (_Float16*)alloc(SZ_TOK * 2);
  float*    W0   = (float*)alloc(SZ_TOK * 4);
  _Float16* h16  = (_Float16*)alloc(SZ_FF * 2);
  _Float16* kvT  = (_Float16*)alloc((size_t)B_SZ * NHEAD * DHEAD * DHEAD * 2);
  float*    ks   = (float*)alloc((size_t)B_SZ * E_DIM * 4);
  float*    den  = (float*)alloc((size_t)B_SZ * N_TOK * NHEAD * 4);
  float*    gx   = (float*)alloc((size_t)B_SZ * E_DIM * 4);
  float*    nx   = (float*)alloc((size_t)B_SZ * E_DIM * 4);
  _Float16* attn16 = Qf;    // reuse (free after q projection)
  _Float16* fn16   = Kf;    // reuse (free after k projection)
  float*    Ftok   = tmp32; // reuse for final token buffer

  // 1) weights -> f16 (native [out,in] layout == col-major B operand)
  k_w2h<<<(int)((SZ_EE + 255) / 256), 256, 0, stream>>>(q_w, Wq, (int)SZ_EE);
  k_w2h<<<(int)((SZ_EE + 255) / 256), 256, 0, stream>>>(k_w, Wk, (int)SZ_EE);
  k_w2h<<<(int)((SZ_EE + 255) / 256), 256, 0, stream>>>(v_w, Wv, (int)SZ_EE);
  k_w2h<<<(int)((SZ_EE + 255) / 256), 256, 0, stream>>>(o_w, Wo, (int)SZ_EE);
  k_w2h<<<(int)((SZ_EE * 4 + 255) / 256), 256, 0, stream>>>(w1, W1, (int)(SZ_EE * 4));
  k_w2h<<<(int)((SZ_EE * 4 + 255) / 256), 256, 0, stream>>>(w2, W2, (int)(SZ_EE * 4));

  // 2) image -> token layout
  dim3 tgrid(N_TOK / 32, E_DIM / 32, B_SZ);
  k_img2tok_f32<<<tgrid, 256, 0, stream>>>(q_img, X);
  k_img2tok_f16<<<tgrid, 256, 0, stream>>>(k_img, Kf);
  k_img2tok_f16<<<tgrid, 256, 0, stream>>>(v_img, Vf);

  // 3) GRN1 -> Qf (f16 tokens)
  k_grn_gx<<<8, 256, 0, stream>>>(X, gx);
  k_grn_nx<<<B_SZ, 256, 0, stream>>>(gx, nx);
  k_grn_apply<<<(int)(SZ_TOK / 256), 256, 0, stream>>>(X, nx, g1g, g1b, Qf);

  // 4) projections
  launch_gemm<0>(stream, Qf, Wq, tmp32, nullptr, q_b, nullptr, nullptr, nullptr,
                 MTOT, E_DIM, E_DIM, E_DIM, E_DIM, E_DIM, 0, 0, 0, 0, 0, 0, 1, 1);
  k_ln<<<MTOT / 8, 256, 0, stream>>>(tmp32, lnq_w, lnq_b, q16, 1);
  launch_gemm<0>(stream, Kf, Wk, tmp32, nullptr, k_b, nullptr, nullptr, nullptr,
                 MTOT, E_DIM, E_DIM, E_DIM, E_DIM, E_DIM, 0, 0, 0, 0, 0, 0, 1, 1);
  k_ln<<<MTOT / 8, 256, 0, stream>>>(tmp32, lnk_w, lnk_b, k16, 1);
  launch_gemm<1>(stream, Vf, Wv, nullptr, v16, v_b, nullptr, nullptr, nullptr,
                 MTOT, E_DIM, E_DIM, E_DIM, E_DIM, E_DIM, 0, 0, 0, 0, 0, 0, 1, 1);

  // 5) linear attention
  k_ksum<<<8, 256, 0, stream>>>(k16, ks);
  k_kv<<<B_SZ * NHEAD, 128, 0, stream>>>(k16, v16, kvT);
  k_den<<<(int)(((size_t)B_SZ * N_TOK * NHEAD) / 256), 256, 0, stream>>>(q16, ks, den);
  // num/den: batched GEMM, z = b*NHEAD + h
  launch_gemm<2>(stream, q16, kvT, nullptr, attn16, nullptr, nullptr, nullptr, den,
                 N_TOK, DHEAD, DHEAD, E_DIM, DHEAD, E_DIM,
                 (long long)N_TOK * E_DIM, DHEAD,
                 (long long)NHEAD * DHEAD * DHEAD, (long long)DHEAD * DHEAD,
                 (long long)N_TOK * E_DIM, DHEAD,
                 NHEAD, B_SZ * NHEAD);
  // o-projection fused with residual: W0 = X + (attn @ Wo^T + o_b) * attn_scalar
  launch_gemm<3>(stream, attn16, Wo, W0, nullptr, o_b, attn_s, X, nullptr,
                 MTOT, E_DIM, E_DIM, E_DIM, E_DIM, E_DIM, 0, 0, 0, 0, 0, 0, 1, 1);

  // 6) GRN2 -> fn16
  k_grn_gx<<<8, 256, 0, stream>>>(W0, gx);
  k_grn_nx<<<B_SZ, 256, 0, stream>>>(gx, nx);
  k_grn_apply<<<(int)(SZ_TOK / 256), 256, 0, stream>>>(W0, nx, g2g, g2b, fn16);

  // 7) FFN
  launch_gemm<4>(stream, fn16, W1, nullptr, h16, b1, nullptr, nullptr, nullptr,
                 MTOT, FF_DIM, E_DIM, E_DIM, E_DIM, FF_DIM, 0, 0, 0, 0, 0, 0, 1, 1);
  launch_gemm<3>(stream, h16, W2, Ftok, nullptr, b2, ffn_s, W0, nullptr,
                 MTOT, E_DIM, FF_DIM, FF_DIM, FF_DIM, E_DIM, 0, 0, 0, 0, 0, 0, 1, 1);

  // 8) final: out = query_img + working * final_scalar  (token -> image)
  k_final<<<tgrid, 256, 0, stream>>>(Ftok, q_img, fin_s, out);
}
